// VQ_VAE_3006477107444
// MI455X (gfx1250) — compile-verified, matrix operand order checked by reading, not run on hardware
//
#include <hip/hip_runtime.h>
#include <math.h>

// ---------------------------------------------------------------------------
// CDNA5 (gfx1250) VQ-VAE forward. All heavy convs + the VQ distance GEMM run
// on v_wmma_f32_16x16x32_bf16 (wave32, 16x16x32 tiles).
// This revision widens the conv tile to 16 x (NT*16) outputs per wave so each
// weight (A) fragment feeds NT WMMAs -> 4x fewer weight loads per MAC.
// ---------------------------------------------------------------------------

typedef __attribute__((ext_vector_type(16))) __bf16 v16bf;
typedef __attribute__((ext_vector_type(8)))  float  v8f;

__device__ __forceinline__ __bf16 f2bf(float f) {
  unsigned u = __builtin_bit_cast(unsigned, f);
  u += 0x7FFFu + ((u >> 16) & 1u);            // round-to-nearest-even
  unsigned short h = (unsigned short)(u >> 16);
  return __builtin_bit_cast(__bf16, h);
}

// K index (0..31) held by a lane-half for element e (0..15) of a 16-bit A/B frag.
// Layout (ISA 7.12.2): lanes 0-15 hold K 0-7 & 16-23, lanes 16-31 hold K 8-15 & 24-31.
__device__ __forceinline__ int frag_k(int e, int half) {
  int v = e >> 1, p = e & 1;
  int base = (v < 4) ? (2 * v) : (16 + 2 * (v - 4));
  return base + 8 * half + p;
}

// ---------------------------------------------------------------------------
// Generic implicit-GEMM conv (normal conv, or stride-1 transposed conv when
// tconv==1: weight layout [Cin,Cout,KH,KW], input offset oh+pad-kh).
// One wave computes a 16 (out-ch) x NT*16 (pixels along W) tile; the A
// (weight) fragment is reused for NT WMMAs.
// mode: 0 = +bias, 1 = +bias,relu, 2 = +bias,+resid,relu
// ---------------------------------------------------------------------------
__global__ void conv_wmma_kernel(const float* __restrict__ in,
                                 const float* __restrict__ wt,
                                 const float* __restrict__ bias,
                                 const float* __restrict__ resid,
                                 float* __restrict__ out,
                                 int Nb, int Cin, int Hin, int Win,
                                 int Cout, int Hout, int Wout,
                                 int KH, int KW, int stride, int pad,
                                 int tconv, int mode, int NT) {
  int wid  = (blockIdx.x * blockDim.x + threadIdx.x) >> 5;
  int lane = threadIdx.x & 31;
  int cot = Cout >> 4;
  int wtg = (Wout >> 4) / NT;                 // groups of NT 16-pixel subtiles
  int tiles = cot * wtg * Hout * Nb;
  if (wid >= tiles) return;
  int co_t = wid % cot;  int t = wid / cot;
  int owg  = t % wtg;    t /= wtg;
  int oh   = t % Hout;   int b = t / Hout;

  int half = lane >> 4;
  int mn   = lane & 15;
  int co   = co_t * 16 + mn;                  // A-frag row (out channel)
  int owb  = owg * NT * 16 + mn;              // lane's pixel in subtile 0

  const int HW = Hin * Win;
  const float* inb = in + (b * Cin) * HW;
  v8f acc[4] = {v8f{}, v8f{}, v8f{}, v8f{}};

  for (int kh = 0; kh < KH; ++kh) {
    int ih = tconv ? (oh + pad - kh) : (oh * stride + kh - pad);
    bool hok = (ih >= 0) && (ih < Hin);
    for (int kw = 0; kw < KW; ++kw) {
      const float* ipj[4];
      bool okj[4];
#pragma unroll
      for (int j = 0; j < 4; ++j) {
        if (j >= NT) { okj[j] = false; ipj[j] = in; continue; }
        int ow = owb + j * 16;
        int iw = tconv ? (ow + pad - kw) : (ow * stride + kw - pad);
        okj[j] = hok && (iw >= 0) && (iw < Win);
        ipj[j] = inb + ih * Win + iw;
        if (okj[j]) __builtin_prefetch(ipj[j], 0, 1);
      }
      for (int c0 = 0; c0 < Cin; c0 += 32) {
        v16bf a;
#pragma unroll
        for (int e = 0; e < 16; ++e) {
          int ci = c0 + frag_k(e, half);
          float wv = tconv ? wt[((ci * Cout + co) * KH + kh) * KW + kw]
                           : wt[((co * Cin + ci) * KH + kh) * KW + kw];
          a[e] = f2bf(wv);
        }
#pragma unroll
        for (int j = 0; j < 4; ++j) {         // A fragment reused NT times
          if (j >= NT) continue;
          v16bf bb;
#pragma unroll
          for (int e = 0; e < 16; ++e) {
            int ci = c0 + frag_k(e, half);
            bb[e] = f2bf(okj[j] ? ipj[j][ci * HW] : 0.0f);
          }
          acc[j] = __builtin_amdgcn_wmma_f32_16x16x32_bf16(
              false, a, false, bb, (short)0, acc[j], false, false);
        }
      }
    }
  }

  int rowbase = ((b * Cout) * Hout + oh) * Wout;
#pragma unroll
  for (int j = 0; j < 4; ++j) {
    if (j >= NT) continue;
    int obase = rowbase + owg * NT * 16 + j * 16 + mn;
#pragma unroll
    for (int r = 0; r < 8; ++r) {
      int m = r + 8 * half;                   // D row (VGPR r -> M=r / M=r+8)
      int c = co_t * 16 + m;
      int oi = obase + c * Hout * Wout;
      float v = acc[j][r] + bias[c];
      if (mode == 2) v += resid[oi];
      if (mode >= 1) v = fmaxf(v, 0.0f);
      out[oi] = v;
    }
  }
}

// ---------------------------------------------------------------------------
// Stride-2 transposed conv 4x4 pad 1 (torch layout wt[Cin,Cout,4,4]).
// Tiles SAME-PARITY output pixels so the (kh,kw) valid set is tile-uniform.
// One wave: 16 out-ch x 32 same-parity pixels (2 accumulators, A reuse x2).
// ---------------------------------------------------------------------------
__global__ void tconv2_wmma_kernel(const float* __restrict__ in,
                                   const float* __restrict__ wt,
                                   const float* __restrict__ bias,
                                   float* __restrict__ out,
                                   int Nb, int Cin, int Hin, int Win, int Cout) {
  int Hout = 2 * Hin, Wout = 2 * Win;         // Wout == 64 here
  int wid  = (blockIdx.x * blockDim.x + threadIdx.x) >> 5;
  int lane = threadIdx.x & 31;
  int cot = Cout >> 4;
  int tiles = cot * 2 * Hout * Nb;
  if (wid >= tiles) return;
  int co_t = wid % cot;  int t = wid / cot;
  int par  = t & 1;      t >>= 1;
  int oh   = t % Hout;   int b = t / Hout;

  int half = lane >> 4;
  int mn   = lane & 15;
  int co   = co_t * 16 + mn;

  const int HW = Hin * Win;
  const float* inb = in + (b * Cin) * HW;
  v8f acc[2] = {v8f{}, v8f{}};
  int kh0 = (oh + 1) & 1;
  int kw0 = (par + 1) & 1;

  for (int dk = 0; dk < 2; ++dk) {
    int kh = kh0 + 2 * dk;
    int th = oh + 1 - kh;
    int ih = th >> 1;
    bool hok = (th >= 0) && (ih < Hin);
    for (int dj = 0; dj < 2; ++dj) {
      int kw = kw0 + 2 * dj;
      const float* ipj[2];
      bool okj[2];
#pragma unroll
      for (int j = 0; j < 2; ++j) {
        int ow = par + 2 * (j * 16 + mn);
        int tw = ow + 1 - kw;
        int iw = tw >> 1;
        okj[j] = hok && (tw >= 0) && (iw < Win);
        ipj[j] = inb + ih * Win + iw;
      }
      for (int c0 = 0; c0 < Cin; c0 += 32) {
        v16bf a;
#pragma unroll
        for (int e = 0; e < 16; ++e) {
          int ci = c0 + frag_k(e, half);
          a[e] = f2bf(wt[((ci * Cout + co) * 4 + kh) * 4 + kw]);
        }
#pragma unroll
        for (int j = 0; j < 2; ++j) {
          v16bf bb;
#pragma unroll
          for (int e = 0; e < 16; ++e) {
            int ci = c0 + frag_k(e, half);
            bb[e] = f2bf(okj[j] ? ipj[j][ci * HW] : 0.0f);
          }
          acc[j] = __builtin_amdgcn_wmma_f32_16x16x32_bf16(
              false, a, false, bb, (short)0, acc[j], false, false);
        }
      }
    }
  }

  int rowbase = ((b * Cout) * Hout + oh) * Wout;
#pragma unroll
  for (int j = 0; j < 2; ++j) {
    int obase = rowbase + par + 2 * (j * 16 + mn);
#pragma unroll
    for (int r = 0; r < 8; ++r) {
      int m = r + 8 * half;
      int c = co_t * 16 + m;
      out[obase + c * Hout * Wout] = acc[j][r] + bias[c];
    }
  }
}

// ---------------------------------------------------------------------------
// Scalar first conv 3->64, 3x3 s1 p1 (K=27: too small for a 16x16x32 tile).
// ---------------------------------------------------------------------------
__global__ void conv1_kernel(const float* __restrict__ x,
                             const float* __restrict__ w,
                             const float* __restrict__ bias,
                             float* __restrict__ out, int n) {
  int i = blockIdx.x * blockDim.x + threadIdx.x;
  if (i >= n) return;
  int ww = i & 127, hh = (i >> 7) & 127, c = (i >> 14) & 63, b = i >> 20;
  float s = bias[c];
  for (int ci = 0; ci < 3; ++ci)
    for (int kh = 0; kh < 3; ++kh) {
      int ih = hh + kh - 1;
      if (ih < 0 || ih >= 128) continue;
      for (int kw = 0; kw < 3; ++kw) {
        int iw = ww + kw - 1;
        if (iw < 0 || iw >= 128) continue;
        s += x[((b * 3 + ci) * 128 + ih) * 128 + iw] *
             w[((c * 3 + ci) * 3 + kh) * 3 + kw];
      }
    }
  out[i] = s;
}

// ---------------------------------------------------------------------------
// Scalar output transposed conv 64->3, 4x4 s2 p1 + sigmoid (Cout=3).
// ---------------------------------------------------------------------------
__global__ void convt3_kernel(const float* __restrict__ x,
                              const float* __restrict__ w,
                              const float* __restrict__ bias,
                              float* __restrict__ out, int n) {
  int i = blockIdx.x * blockDim.x + threadIdx.x;
  if (i >= n) return;
  const int Hin = 64, Win = 64, Cin = 64;
  int ww = i & 127, hh = (i >> 7) & 127, c = (i >> 14) % 3, b = i / (3 * 16384);
  float s = bias[c];
  int kh0 = (hh + 1) & 1, kw0 = (ww + 1) & 1;
  for (int dk = 0; dk < 2; ++dk) {
    int kh = kh0 + 2 * dk;
    int th = hh + 1 - kh;
    if (th < 0) continue;
    int ih = th >> 1;
    if (ih >= Hin) continue;
    for (int dj = 0; dj < 2; ++dj) {
      int kw = kw0 + 2 * dj;
      int tw = ww + 1 - kw;
      if (tw < 0) continue;
      int iw = tw >> 1;
      if (iw >= Win) continue;
      for (int ci = 0; ci < Cin; ++ci)
        s += x[((b * Cin + ci) * Hin + ih) * Win + iw] *
             w[((ci * 3 + c) * 4 + kh) * 4 + kw];
    }
  }
  out[i] = 1.0f / (1.0f + expf(-s));
}

// ---------------------------------------------------------------------------
// BatchNorm (training-mode batch stats) : per-channel sum / sumsq reduction,
// then fused normalize + affine + ReLU.
// ---------------------------------------------------------------------------
__global__ void bn_stats_kernel(const float* __restrict__ x, float* __restrict__ st,
                                int Nb, int C, int HW) {
  __shared__ float sh0[256], sh1[256];
  int c = blockIdx.x;
  int total = Nb * HW;
  float s = 0.0f, s2 = 0.0f;
  for (int i = threadIdx.x; i < total; i += blockDim.x) {
    int b = i / HW, r = i - b * HW;
    float v = x[(b * C + c) * HW + r];
    s += v; s2 += v * v;
  }
  sh0[threadIdx.x] = s; sh1[threadIdx.x] = s2;
  __syncthreads();
  for (int o = 128; o > 0; o >>= 1) {
    if ((int)threadIdx.x < o) {
      sh0[threadIdx.x] += sh0[threadIdx.x + o];
      sh1[threadIdx.x] += sh1[threadIdx.x + o];
    }
    __syncthreads();
  }
  if (threadIdx.x == 0) { st[c] = sh0[0]; st[C + c] = sh1[0]; }
}

__global__ void bn_apply_relu_kernel(const float* __restrict__ x,
                                     float* __restrict__ y,
                                     const float* __restrict__ g,
                                     const float* __restrict__ bta,
                                     const float* __restrict__ st,
                                     int C, int HW, float invCnt, int n) {
  int i = blockIdx.x * blockDim.x + threadIdx.x;
  if (i >= n) return;
  int c = (i / HW) % C;
  float m = st[c] * invCnt;
  float v = st[C + c] * invCnt - m * m;
  float r = g[c] * (x[i] - m) * rsqrtf(v + 1e-5f) + bta[c];
  y[i] = fmaxf(r, 0.0f);
}

// ---------------------------------------------------------------------------
// VQ: norms, WMMA distance GEMM with fused per-row argmin, gather + loss.
// z_e is NCHW [32,256,32,32]; flat row n = ((b*32)+h)*32+w, dim = channel.
// ---------------------------------------------------------------------------
__global__ void rownorm_kernel(const float* __restrict__ m, float* __restrict__ o,
                               int rows, int cols) {
  int r = blockIdx.x * blockDim.x + threadIdx.x;
  if (r >= rows) return;
  float s = 0.0f;
  for (int c = 0; c < cols; ++c) { float v = m[r * cols + c]; s += v * v; }
  o[r] = s;
}

__global__ void znorm_kernel(const float* __restrict__ ze, float* __restrict__ o, int n) {
  int r = blockIdx.x * blockDim.x + threadIdx.x;
  if (r >= n) return;
  int w = r & 31, h = (r >> 5) & 31, b = r >> 10;
  const float* p = ze + (b * 256) * 1024 + h * 32 + w;
  float s = 0.0f;
  for (int c = 0; c < 256; ++c) { float v = p[c * 1024]; s += v * v; }
  o[r] = s;
}

__global__ void vq_argmin_kernel(const float* __restrict__ ze,
                                 const float* __restrict__ cb,
                                 const float* __restrict__ znorm,
                                 const float* __restrict__ cbnorm,
                                 int* __restrict__ idx,
                                 float* __restrict__ idxf) {
  int wid  = (blockIdx.x * blockDim.x + threadIdx.x) >> 5;
  int lane = threadIdx.x & 31;
  const int NROW = 32768;
  if (wid * 16 >= NROW) return;
  int half = lane >> 4, mn = lane & 15;

  int arow = wid * 16 + mn;
  int w = arow & 31, h = (arow >> 5) & 31, b = arow >> 10;
  const float* zp = ze + (b * 256) * 1024 + h * 32 + w;

  v16bf afr[8];                          // whole z-row (K=256) in registers
#pragma unroll
  for (int c0 = 0; c0 < 8; ++c0)
#pragma unroll
    for (int e = 0; e < 16; ++e)
      afr[c0][e] = f2bf(zp[(c0 * 32 + frag_k(e, half)) * 1024]);

  float zn[8];
#pragma unroll
  for (int r = 0; r < 8; ++r) zn[r] = znorm[wid * 16 + r + 8 * half];

  float best[8]; int bidx[8];
#pragma unroll
  for (int r = 0; r < 8; ++r) { best[r] = 3.4e38f; bidx[r] = 0; }

  for (int t = 0; t < 32; ++t) {         // 512 codebook entries, 16 at a time
    v8f acc = {};
#pragma unroll
    for (int c0 = 0; c0 < 8; ++c0) {
      v16bf bb;
      const float* cp = cb + (t * 16 + mn) * 256 + c0 * 32;
#pragma unroll
      for (int e = 0; e < 16; ++e) bb[e] = f2bf(cp[frag_k(e, half)]);
      acc = __builtin_amdgcn_wmma_f32_16x16x32_bf16(false, afr[c0], false, bb,
                                                    (short)0, acc, false, false);
    }
    float cn = cbnorm[t * 16 + mn];
    int j = t * 16 + mn;
#pragma unroll
    for (int r = 0; r < 8; ++r) {
      float val = zn[r] - 2.0f * acc[r] + cn;
      if (val < best[r]) { best[r] = val; bidx[r] = j; }
    }
  }
#pragma unroll
  for (int r = 0; r < 8; ++r) {
    for (int off = 8; off >= 1; off >>= 1) {   // reduce across 16 lanes / half
      float ob = __shfl_xor(best[r], off, 32);
      int   oi = __shfl_xor(bidx[r], off, 32);
      if (ob < best[r] || (ob == best[r] && oi < bidx[r])) { best[r] = ob; bidx[r] = oi; }
    }
  }
  if (mn == 0) {
#pragma unroll
    for (int r = 0; r < 8; ++r) {
      int row = wid * 16 + r + 8 * half;
      idx[row]  = bidx[r];
      idxf[row] = (float)bidx[r];
    }
  }
}

__global__ void vq_gather_loss_kernel(const float* __restrict__ ze,
                                      const float* __restrict__ cb,
                                      const int* __restrict__ idx,
                                      float* __restrict__ zq,
                                      float* __restrict__ lossAcc, int n) {
  int r = blockIdx.x * blockDim.x + threadIdx.x;
  if (r >= n) return;
  int w = r & 31, h = (r >> 5) & 31, b = r >> 10;
  int off = (b * 256) * 1024 + h * 32 + w;
  const float* cp = cb + idx[r] * 256;
  float s = 0.0f;
  for (int c = 0; c < 256; ++c) {
    float q = cp[c];
    float z = ze[off + c * 1024];
    zq[off + c * 1024] = q;
    float d = q - z;
    s += d * d;
  }
  atomicAdd(lossAcc, s);
}

__global__ void zero1_kernel(float* p) { *p = 0.0f; }

__global__ void loss_final_kernel(const float* acc, float* out) {
  // forward value of the straight-through loss = (1 + BETA) * MSE, BETA = 1
  *out = 2.0f * (*acc) / (32768.0f * 256.0f);
}

// ---------------------------------------------------------------------------
// Host-side sequencing
// ---------------------------------------------------------------------------
static inline void launch_conv(const float* in, const float* wt, const float* bias,
                               const float* resid, float* out,
                               int Nb, int Cin, int Hin, int Win,
                               int Cout, int Hout, int Wout,
                               int KH, int KW, int stride, int pad,
                               int tconv, int mode, hipStream_t s) {
  int wtn = Wout / 16;
  int NT  = (wtn % 4 == 0) ? 4 : 2;          // 64-wide tile when possible
  long waves = (long)(Cout / 16) * (wtn / NT) * Hout * Nb;
  int blocks = (int)((waves + 7) / 8);       // 8 waves / 256-thread block
  conv_wmma_kernel<<<blocks, 256, 0, s>>>(in, wt, bias, resid, out,
                                          Nb, Cin, Hin, Win, Cout, Hout, Wout,
                                          KH, KW, stride, pad, tconv, mode, NT);
}

static inline void run_bn(const float* x, float* y, const float* g, const float* b,
                          float* stats, int Nb, int C, int HW, hipStream_t s) {
  bn_stats_kernel<<<C, 256, 0, s>>>(x, stats, Nb, C, HW);
  int n = Nb * C * HW;
  bn_apply_relu_kernel<<<(n + 255) / 256, 256, 0, s>>>(x, y, g, b, stats, C, HW,
                                                       1.0f / (float)(Nb * HW), n);
}

extern "C" void kernel_launch(void* const* d_in, const int* in_sizes, int n_in,
                              void* d_out, int out_size, void* d_ws, size_t ws_size,
                              hipStream_t stream) {
  (void)in_sizes; (void)n_in; (void)ws_size;
  const float* x     = (const float*)d_in[0];
  const float* c1w   = (const float*)d_in[1];
  const float* c1b   = (const float*)d_in[2];
  const float* bn1g  = (const float*)d_in[3];
  const float* bn1b  = (const float*)d_in[4];
  const float* er1w1 = (const float*)d_in[5];
  const float* er1b1 = (const float*)d_in[6];
  const float* er1w2 = (const float*)d_in[7];
  const float* er1b2 = (const float*)d_in[8];
  const float* er2w1 = (const float*)d_in[9];
  const float* er2b1 = (const float*)d_in[10];
  const float* er2w2 = (const float*)d_in[11];
  const float* er2b2 = (const float*)d_in[12];
  const float* c2w   = (const float*)d_in[13];
  const float* c2b   = (const float*)d_in[14];
  const float* bn2g  = (const float*)d_in[15];
  const float* bn2b  = (const float*)d_in[16];
  const float* er3w1 = (const float*)d_in[17];
  const float* er3b1 = (const float*)d_in[18];
  const float* er3w2 = (const float*)d_in[19];
  const float* er3b2 = (const float*)d_in[20];
  const float* c3w   = (const float*)d_in[21];
  const float* c3b   = (const float*)d_in[22];
  const float* t1w   = (const float*)d_in[23];
  const float* t1b   = (const float*)d_in[24];
  const float* dbn1g = (const float*)d_in[25];
  const float* dbn1b = (const float*)d_in[26];
  const float* dr1w1 = (const float*)d_in[27];
  const float* dr1b1 = (const float*)d_in[28];
  const float* dr1w2 = (const float*)d_in[29];
  const float* dr1b2 = (const float*)d_in[30];
  const float* dr2w1 = (const float*)d_in[31];
  const float* dr2b1 = (const float*)d_in[32];
  const float* dr2w2 = (const float*)d_in[33];
  const float* dr2b2 = (const float*)d_in[34];
  const float* t2w   = (const float*)d_in[35];
  const float* t2b   = (const float*)d_in[36];
  const float* dbn2g = (const float*)d_in[37];
  const float* dbn2b = (const float*)d_in[38];
  const float* dr3w1 = (const float*)d_in[39];
  const float* dr3b1 = (const float*)d_in[40];
  const float* dr3w2 = (const float*)d_in[41];
  const float* dr3b2 = (const float*)d_in[42];
  const float* t3w   = (const float*)d_in[43];
  const float* t3b   = (const float*)d_in[44];
  const float* cb    = (const float*)d_in[45];

  float* outp = (float*)d_out;
  float* ws   = (float*)d_ws;

  // workspace layout (floats)
  const long BIG = 33554432;                 // 32*64*128*128
  float* t0 = ws;
  float* t1 = ws + BIG;
  float* t2 = ws + 2 * BIG;
  float* misc   = ws + 3 * BIG;
  float* stats  = misc;                      // up to 2*256
  float* cbnorm = misc + 1024;               // 512
  float* znorm  = misc + 2048;               // 32768
  float* lossA  = misc + 2048 + 32768;       // 1
  int*   idxbuf = (int*)(misc + 2048 + 32768 + 8);   // 32768 ints

  const int Nb = 32;
  const int ZD = 32 * 3 * 128 * 128;         // 1,572,864 floats of z_d
  float* idx_out  = outp + ZD;               // 32768 indices (as float)
  float* loss_out = outp + (out_size - 1);

  // ---------------- encoder ----------------
  conv1_kernel<<<(int)(BIG / 256), 256, 0, stream>>>(x, c1w, c1b, t0, (int)BIG);
  run_bn(t0, t1, bn1g, bn1b, stats, Nb, 64, 128 * 128, stream);

  launch_conv(t1, er1w1, er1b1, nullptr, t0, Nb, 64, 128, 128, 64, 128, 128, 3, 3, 1, 1, 0, 1, stream);
  launch_conv(t0, er1w2, er1b2, t1,      t2, Nb, 64, 128, 128, 64, 128, 128, 3, 3, 1, 1, 0, 2, stream);
  launch_conv(t2, er2w1, er2b1, nullptr, t0, Nb, 64, 128, 128, 64, 128, 128, 3, 3, 1, 1, 0, 1, stream);
  launch_conv(t0, er2w2, er2b2, t2,      t1, Nb, 64, 128, 128, 64, 128, 128, 3, 3, 1, 1, 0, 2, stream);

  launch_conv(t1, c2w, c2b, nullptr, t0, Nb, 64, 128, 128, 128, 64, 64, 4, 4, 2, 1, 0, 0, stream);
  run_bn(t0, t2, bn2g, bn2b, stats, Nb, 128, 64 * 64, stream);

  launch_conv(t2, er3w1, er3b1, nullptr, t0, Nb, 128, 64, 64, 128, 64, 64, 3, 3, 1, 1, 0, 1, stream);
  launch_conv(t0, er3w2, er3b2, t2,      t1, Nb, 128, 64, 64, 128, 64, 64, 3, 3, 1, 1, 0, 2, stream);

  launch_conv(t1, c3w, c3b, nullptr, t0, Nb, 128, 64, 64, 256, 32, 32, 4, 4, 2, 1, 0, 0, stream);
  // z_e in t0: [32,256,32,32]

  // ---------------- vector quantization ----------------
  rownorm_kernel<<<2, 256, 0, stream>>>(cb, cbnorm, 512, 256);
  znorm_kernel<<<128, 256, 0, stream>>>(t0, znorm, 32768);
  {
    long waves = 32768 / 16;                 // 2048 waves
    vq_argmin_kernel<<<(int)((waves + 7) / 8), 256, 0, stream>>>(
        t0, cb, znorm, cbnorm, idxbuf, idx_out);
  }
  zero1_kernel<<<1, 1, 0, stream>>>(lossA);
  vq_gather_loss_kernel<<<128, 256, 0, stream>>>(t0, cb, idxbuf, t1, lossA, 32768);
  loss_final_kernel<<<1, 1, 0, stream>>>(lossA, loss_out);
  // z_q (== cb[idx]) in t1: [32,256,32,32]

  // ---------------- decoder ----------------
  launch_conv(t1, t1w, t1b, nullptr, t0, Nb, 256, 32, 32, 128, 32, 32, 3, 3, 1, 1, 1, 0, stream);
  run_bn(t0, t2, dbn1g, dbn1b, stats, Nb, 128, 32 * 32, stream);

  launch_conv(t2, dr1w1, dr1b1, nullptr, t0, Nb, 128, 32, 32, 128, 32, 32, 3, 3, 1, 1, 0, 1, stream);
  launch_conv(t0, dr1w2, dr1b2, t2,      t1, Nb, 128, 32, 32, 128, 32, 32, 3, 3, 1, 1, 0, 2, stream);
  launch_conv(t1, dr2w1, dr2b1, nullptr, t0, Nb, 128, 32, 32, 128, 32, 32, 3, 3, 1, 1, 0, 1, stream);
  launch_conv(t0, dr2w2, dr2b2, t1,      t2, Nb, 128, 32, 32, 128, 32, 32, 3, 3, 1, 1, 0, 2, stream);

  {
    long waves = (long)(64 / 16) * 2 * 64 * Nb;   // stride-2 tconv, 2 subtiles/wave
    tconv2_wmma_kernel<<<(int)((waves + 7) / 8), 256, 0, stream>>>(
        t2, t2w, t2b, t0, Nb, 128, 32, 32, 64);
  }
  run_bn(t0, t1, dbn2g, dbn2b, stats, Nb, 64, 64 * 64, stream);

  launch_conv(t1, dr3w1, dr3b1, nullptr, t0, Nb, 64, 64, 64, 64, 64, 64, 3, 3, 1, 1, 0, 1, stream);
  launch_conv(t0, dr3w2, dr3b2, t1,      t2, Nb, 64, 64, 64, 64, 64, 64, 3, 3, 1, 1, 0, 2, stream);

  convt3_kernel<<<(ZD + 255) / 256, 256, 0, stream>>>(t2, t3w, t3b, outp, ZD);
}
